// NonLocalBlock_24240795419317
// MI455X (gfx1250) — matrix operand order repaired and checked by reading, not compiled, not run on hardware
//
#include <hip/hip_runtime.h>
#include <cstdint>
#include <cstddef>

typedef __bf16 bf16_t;
typedef __attribute__((ext_vector_type(16))) __bf16 v16bf;
typedef __attribute__((ext_vector_type(8)))  float  v8f;
typedef __attribute__((ext_vector_type(4)))  unsigned int v4u;
typedef int v4i_t __attribute__((vector_size(16)));   // gcc-style, matches builtin params

#ifndef __has_builtin
#define __has_builtin(x) 0
#endif

#if __has_builtin(__builtin_amdgcn_global_load_async_to_lds_b128) && \
    __has_builtin(__builtin_amdgcn_s_wait_asynccnt)
#define HAVE_ASYNC_LDS 1
#else
#define HAVE_ASYNC_LDS 0
#endif

#if __has_builtin(__builtin_amdgcn_tensor_load_to_lds) && \
    __has_builtin(__builtin_amdgcn_s_wait_tensorcnt)
#define HAVE_TDM 1
#else
#define HAVE_TDM 0
#endif

#if HAVE_ASYNC_LDS || HAVE_TDM
#define GLOBAL_AS __attribute__((address_space(1)))
#define LDS_AS    __attribute__((address_space(3)))
#endif

union FragAB { v4u u[2]; v16bf v; };

__device__ __forceinline__ v8f v8f_zero() {
  v8f z = {0.f, 0.f, 0.f, 0.f, 0.f, 0.f, 0.f, 0.f};
  return z;
}

// D = A*B + C, bf16 in, f32 accumulate
__device__ __forceinline__ v8f wmma_bf16(v16bf a, v16bf b, v8f c) {
  return __builtin_amdgcn_wmma_f32_16x16x32_bf16(
      false, a, false, b, (short)0, c, false, false);
}

// A-fragment (16x32 bf16): lane m = lane%16 (absolute row passed in), half = lane/16.
// K = kt*32 + {half*8+0..7, 16+half*8+0..7} -> two contiguous 16B chunks of the row.
__device__ __forceinline__ v16bf load_a_frag(const bf16_t* __restrict__ base,
                                             int row, int row_u4, int kt, int hf) {
  const v4u* p = (const v4u*)base;
  int i0 = row * row_u4 + kt * 4 + hf;
  FragAB f;
  f.u[0] = p[i0];
  f.u[1] = p[i0 + 2];
  return f.v;
}

// B-fragment (32x16 bf16) from an LDS tile in fragment order:
// tile[n*32 + kw] holds (K=kw, N=n); lane n = lane%16, K = {half*8+0..7, 16+half*8+0..7}.
__device__ __forceinline__ v16bf load_b_frag(const bf16_t* tile, int n, int hf) {
  const v4u* p = (const v4u*)(tile + n * 32);
  FragAB f;
  f.u[0] = p[hf];
  f.u[1] = p[hf + 2];
  return f.v;
}

// 16-byte global -> LDS copy (async-to-LDS when available)
__device__ __forceinline__ void copy16_g2l(bf16_t* lds_dst, const bf16_t* gsrc) {
#if HAVE_ASYNC_LDS
  __builtin_amdgcn_global_load_async_to_lds_b128(
      (GLOBAL_AS v4i_t*)gsrc, (LDS_AS v4i_t*)lds_dst, 0, 0);
#else
  *(v4u*)lds_dst = *(const v4u*)gsrc;
#endif
}
__device__ __forceinline__ void stage_fence_async() {
#if HAVE_ASYNC_LDS
  __builtin_amdgcn_s_wait_asynccnt(0);
#endif
}

#if HAVE_TDM
typedef unsigned int u32x4 __attribute__((ext_vector_type(4)));
typedef int i32x4 __attribute__((ext_vector_type(4)));
typedef int i32x8 __attribute__((ext_vector_type(8)));

__device__ __forceinline__ unsigned lds_off(const void* p) {
  return (unsigned)(unsigned long long)(const LDS_AS char*)p;
}

// 1-D contiguous TDM copy, data_size = 8B units.
__device__ __forceinline__ void tdm_load_1d(bf16_t* lds_dst, const bf16_t* gsrc,
                                            unsigned nbytes) {
  unsigned long long ga = (unsigned long long)gsrc;
  unsigned n8 = nbytes >> 3;  // 8-byte elements; fits tile_dim0 (16b)
  u32x4 g0 = {1u /*count=1*/, lds_off(lds_dst),
              (unsigned)ga,
              (unsigned)((ga >> 32) & 0x01ffffffu) | 0x80000000u /*type=2*/};
  i32x8 g1 = {(int)0x00030000 /*data_size=3(8B)*/,
              (int)(n8 << 16)                    /*tensor_dim0[15:0]*/,
              (int)((n8 >> 16) | 0x10000u)       /*tensor_dim0[31:16], tensor_dim1=1*/,
              (int)(n8 << 16)                    /*tile_dim0*/,
              0 /*tile_dim1,2 = 0*/,
              (int)n8 /*tensor_dim0_stride[31:0]*/, 0, 0};
  i32x4 g2 = {0, 0, 0, 0};
  i32x4 g3 = {0, 0, 0, 0};
#if __clang_major__ >= 23
  i32x8 g4 = {0, 0, 0, 0, 0, 0, 0, 0};
  __builtin_amdgcn_tensor_load_to_lds(g0, g1, g2, g3, g4, 0);
#else
  __builtin_amdgcn_tensor_load_to_lds(g0, g1, g2, g3, 0);
#endif
}
#endif  // HAVE_TDM

// ---------------------------------------------------------------------------
// f32 -> bf16 plain convert (weights)
// ---------------------------------------------------------------------------
__global__ void cvt_bf16_kernel(const float* __restrict__ in,
                                bf16_t* __restrict__ out, int n) {
  int i = blockIdx.x * blockDim.x + threadIdx.x;
  int stride = gridDim.x * blockDim.x;
  for (; i < n; i += stride) out[i] = (bf16_t)in[i];
}

// x[b,c,hw] f32 -> x_sw bf16, tile layout [(b*16+hwt)*8+kt][n][kw]
__global__ void cvt_x_swz_kernel(const float* __restrict__ in,
                                 bf16_t* __restrict__ out) {
  int f = blockIdx.x * blockDim.x + threadIdx.x;  // [0, 8*256*1024)
  int b = f >> 18;
  int c = (f >> 10) & 255;
  int hw = f & 1023;
  int idx = (((b * 16 + (hw >> 6)) * 8 + (c >> 5)) << 11) + ((hw & 63) << 5) + (c & 31);
  out[idx] = (bf16_t)in[f];
}

// ---------------------------------------------------------------------------
// QKV: T/P/G[b,o,hw] = sum_c W[o,c]*x[b,c,hw] + bias[o]
// M=128 (8 waves x 16), N=64, K=256 (8 k-tiles), double-buffered X tiles.
// ---------------------------------------------------------------------------
__global__ __launch_bounds__(256) void qkv_kernel(
    const bf16_t* __restrict__ xs,                       // swizzled x
    const bf16_t* __restrict__ tw, const bf16_t* __restrict__ pw,
    const bf16_t* __restrict__ gw,                       // [128,256]
    const float* __restrict__ tbs, const float* __restrict__ pbs,
    const float* __restrict__ gbs,
    bf16_t* __restrict__ Tm,                             // [8192,128] row-major
    bf16_t* __restrict__ Psw, bf16_t* __restrict__ Gsw) {
  __shared__ alignas(16) bf16_t Xl[2][64 * 32];

  const int tid  = threadIdx.x;
  const int lane = tid & 31;
  const int wave = tid >> 5;
  const int hf   = lane >> 4;
  const int ln   = lane & 15;
  const int b    = blockIdx.x >> 4;
  const int hwt  = blockIdx.x & 15;
  const int hw0  = hwt * 64;
  const int mrow = wave * 16 + ln;
  const size_t tbase = (size_t)(b * 16 + hwt) * 8;

  v8f acc[3][4];
#pragma unroll
  for (int s = 0; s < 3; ++s)
#pragma unroll
    for (int nt = 0; nt < 4; ++nt) acc[s][nt] = v8f_zero();

  copy16_g2l(Xl[0] + tid * 8, xs + (tbase << 11) + tid * 8);
  stage_fence_async();
  __syncthreads();

  for (int kt = 0; kt < 8; ++kt) {
    const int cur = kt & 1;
    if (kt + 1 < 8)  // issue next tile, no wait yet
      copy16_g2l(Xl[cur ^ 1] + tid * 8,
                 xs + ((tbase + kt + 1) << 11) + tid * 8);

    v16bf at = load_a_frag(tw, mrow, 32, kt, hf);
    v16bf ap = load_a_frag(pw, mrow, 32, kt, hf);
    v16bf ag = load_a_frag(gw, mrow, 32, kt, hf);
#pragma unroll
    for (int nt = 0; nt < 4; ++nt) {
      v16bf bx = load_b_frag(Xl[cur], nt * 16 + ln, hf);
      acc[0][nt] = wmma_bf16(at, bx, acc[0][nt]);
      acc[1][nt] = wmma_bf16(ap, bx, acc[1][nt]);
      acc[2][nt] = wmma_bf16(ag, bx, acc[2][nt]);
    }

    stage_fence_async();
    __syncthreads();
  }

#pragma unroll
  for (int r = 0; r < 8; ++r) {
    int o = wave * 16 + r + 8 * hf;  // D-layout row
    float bt = tbs[o], bp = pbs[o], bg = gbs[o];
#pragma unroll
    for (int nt = 0; nt < 4; ++nt) {
      int hw = hw0 + nt * 16 + ln;
      // theta: row-major [8192,128]
      Tm[((size_t)b * 128 + o) * 1024 + hw] = (bf16_t)(acc[0][nt][r] + bt);
      // phi matrix coords: r_m = b*16 + o/8, c_m = (o%8)*1024 + hw -> swizzle
      {
        int rm = b * 16 + (o >> 3);
        int cm = ((o & 7) << 10) + hw;
        int idx = (((cm >> 6) * 4 + (rm >> 5)) << 11) + ((cm & 63) << 5) + (rm & 31);
        Psw[idx] = (bf16_t)(acc[1][nt][r] + bp);
      }
      // g matrix coords: i = b*1024 + o*8 + hw/128, j = hw%128 -> swizzle
      {
        int im = (b << 10) + (o << 3) + (hw >> 7);
        int jm = hw & 127;
        int idx = (((im >> 6) * 2 + ((im & 63) >> 5)) << 12) + (jm << 5) + (im & 31);
        Gsw[idx] = (bf16_t)(acc[2][nt][r] + bg);
      }
    }
  }
}

// ---------------------------------------------------------------------------
// Flash attention tile staging (TDM preferred, else async, else plain)
// ---------------------------------------------------------------------------
__device__ __forceinline__ void stage_tiles(bf16_t* Pld, bf16_t* Gld,
                                            const bf16_t* __restrict__ Psw,
                                            const bf16_t* __restrict__ Gsw,
                                            int nt0, int tid, int wave) {
#if HAVE_TDM
  (void)tid;
  if (wave == 0) {
    tdm_load_1d(Pld, Psw + ((size_t)nt0 << 13), 16384);
    tdm_load_1d(Gld, Gsw + ((size_t)nt0 << 13), 16384);
  }
#else
  (void)wave;
#pragma unroll
  for (int j = 0; j < 4; ++j) {
    int e = (tid + j * 256) * 8;
    copy16_g2l(Pld + e, Psw + ((size_t)nt0 << 13) + e);
    copy16_g2l(Gld + e, Gsw + ((size_t)nt0 << 13) + e);
  }
#endif
}

__device__ __forceinline__ void stage_wait(int wave) {
#if HAVE_TDM
  if (wave == 0) __builtin_amdgcn_s_wait_tensorcnt(0);
#else
  (void)wave;
  stage_fence_async();
#endif
}

// ---------------------------------------------------------------------------
// Flash attention: O = softmax(T @ P) @ G (8192x8192 S never materialized)
// Block = 128 query rows (8 waves x 16); double-buffered 64-col P/G tiles.
// ---------------------------------------------------------------------------
__global__ __launch_bounds__(256) void flash_kernel(
    const bf16_t* __restrict__ Tm,    // [8192,128]
    const bf16_t* __restrict__ Psw,   // swizzled, 16KB tiles per nt0
    const bf16_t* __restrict__ Gsw,   // swizzled, 16KB tiles per nt0
    bf16_t* __restrict__ Osw) {       // swizzled for outconv
  __shared__ alignas(16) bf16_t Pl[2][4 * 64 * 32];
  __shared__ alignas(16) bf16_t Gl[2][2 * 128 * 32];
  __shared__ alignas(16) bf16_t Pr[8 * 16 * 64];

  const int tid  = threadIdx.x;
  const int lane = tid & 31;
  const int wave = tid >> 5;
  const int hf   = lane >> 4;
  const int ln   = lane & 15;
  const int r0   = blockIdx.x * 128 + wave * 16;

  v16bf qf[4];
#pragma unroll
  for (int kt = 0; kt < 4; ++kt) qf[kt] = load_a_frag(Tm, r0 + ln, 16, kt, hf);

  v8f oacc[8];
#pragma unroll
  for (int ot = 0; ot < 8; ++ot) oacc[ot] = v8f_zero();

  float rmax[8], rsum[8];
#pragma unroll
  for (int r = 0; r < 8; ++r) { rmax[r] = -1e30f; rsum[r] = 0.0f; }

  stage_tiles(Pl[0], Gl[0], Psw, Gsw, 0, tid, wave);
  stage_wait(wave);
  __syncthreads();

  for (int nt0 = 0; nt0 < 128; ++nt0) {
    const int cur = nt0 & 1;
    if (nt0 + 1 < 128)  // overlap next tile's copy with compute
      stage_tiles(Pl[cur ^ 1], Gl[cur ^ 1], Psw, Gsw, nt0 + 1, tid, wave);

    const bf16_t* Plc = Pl[cur];
    const bf16_t* Glc = Gl[cur];

    // ---- S = Q @ P-tile (16x64 per wave) ----
    v8f sacc[4];
#pragma unroll
    for (int nt = 0; nt < 4; ++nt) {
      sacc[nt] = v8f_zero();
#pragma unroll
      for (int kt = 0; kt < 4; ++kt) {
        v16bf bp = load_b_frag(Plc + kt * (64 * 32), nt * 16 + ln, hf);
        sacc[nt] = wmma_bf16(qf[kt], bp, sacc[nt]);
      }
    }

    // ---- online softmax (rows live in 16-lane halves) ----
    float nm[8], psum[8];
#pragma unroll
    for (int r = 0; r < 8; ++r) {
      float t = sacc[0][r];
      t = fmaxf(t, sacc[1][r]);
      t = fmaxf(t, sacc[2][r]);
      t = fmaxf(t, sacc[3][r]);
      t = fmaxf(t, __shfl_xor(t, 1));
      t = fmaxf(t, __shfl_xor(t, 2));
      t = fmaxf(t, __shfl_xor(t, 4));
      t = fmaxf(t, __shfl_xor(t, 8));
      nm[r] = fmaxf(rmax[r], t);
      psum[r] = 0.0f;
    }
#pragma unroll
    for (int nt = 0; nt < 4; ++nt) {
#pragma unroll
      for (int r = 0; r < 8; ++r) {
        float p = __expf(sacc[nt][r] - nm[r]);
        psum[r] += p;
        Pr[wave * 1024 + (r + 8 * hf) * 64 + nt * 16 + ln] = (bf16_t)p;
      }
    }
#pragma unroll
    for (int r = 0; r < 8; ++r) {
      float s = psum[r];
      s += __shfl_xor(s, 1);
      s += __shfl_xor(s, 2);
      s += __shfl_xor(s, 4);
      s += __shfl_xor(s, 8);
      float scale = __expf(rmax[r] - nm[r]);
      rsum[r] = rsum[r] * scale + s;
      rmax[r] = nm[r];
#pragma unroll
      for (int ot = 0; ot < 8; ++ot) oacc[ot][r] *= scale;
    }

    // ---- O += probs @ G-tile ----
    v16bf af[2];
#pragma unroll
    for (int k2 = 0; k2 < 2; ++k2)
      af[k2] = load_a_frag(Pr + wave * 1024, ln, 8, k2, hf);
#pragma unroll
    for (int ot = 0; ot < 8; ++ot) {
#pragma unroll
      for (int k2 = 0; k2 < 2; ++k2) {
        v16bf bg = load_b_frag(Glc + k2 * (128 * 32), ot * 16 + ln, hf);
        oacc[ot] = wmma_bf16(af[k2], bg, oacc[ot]);
      }
    }

    stage_wait(wave);   // next tile resident before the barrier releases it
    __syncthreads();
  }

  // normalize + store into outconv-swizzled layout
#pragma unroll
  for (int r = 0; r < 8; ++r) rsum[r] = 1.0f / rsum[r];
#pragma unroll
  for (int ot = 0; ot < 8; ++ot) {
#pragma unroll
    for (int r = 0; r < 8; ++r) {
      int i = r0 + r + 8 * hf;          // O row
      int j = ot * 16 + ln;             // O col
      int b = i >> 10;
      int o = (i & 1023) >> 3;
      int hw = ((i & 7) << 7) | j;
      int idx = (((b * 16 + (hw >> 6)) * 4 + (o >> 5)) << 11) + ((hw & 63) << 5) + (o & 31);
      Osw[idx] = (bf16_t)(oacc[ot][r] * rsum[r]);
    }
  }
}

// ---------------------------------------------------------------------------
// Output conv: y[b,co,hw] = sum_o Ww[co,o]*O[b,o,hw] + Wb[co] + x[b,co,hw]
// M=256 (2 m-tiles/wave), N=64, K=128 (4 k-tiles), double-buffered O tiles.
// ---------------------------------------------------------------------------
__global__ __launch_bounds__(256) void outconv_kernel(
    const bf16_t* __restrict__ Osw,
    const bf16_t* __restrict__ Wwb,   // [256,128]
    const float* __restrict__ Wbias,
    const float* __restrict__ x,
    float* __restrict__ y) {
  __shared__ alignas(16) bf16_t Ol[2][64 * 32];

  const int tid  = threadIdx.x;
  const int lane = tid & 31;
  const int wave = tid >> 5;
  const int hf   = lane >> 4;
  const int ln   = lane & 15;
  const int b    = blockIdx.x >> 4;
  const int hwt  = blockIdx.x & 15;
  const int hw0  = hwt * 64;
  const size_t tbase = (size_t)(b * 16 + hwt) * 4;

  v8f acc[2][4];
#pragma unroll
  for (int mi = 0; mi < 2; ++mi)
#pragma unroll
    for (int nt = 0; nt < 4; ++nt) acc[mi][nt] = v8f_zero();

  copy16_g2l(Ol[0] + tid * 8, Osw + (tbase << 11) + tid * 8);
  stage_fence_async();
  __syncthreads();

  for (int kt = 0; kt < 4; ++kt) {
    const int cur = kt & 1;
    if (kt + 1 < 4)
      copy16_g2l(Ol[cur ^ 1] + tid * 8,
                 Osw + ((tbase + kt + 1) << 11) + tid * 8);

    v16bf a0 = load_a_frag(Wwb, wave * 16 + ln, 16, kt, hf);
    v16bf a1 = load_a_frag(Wwb, (wave + 8) * 16 + ln, 16, kt, hf);
#pragma unroll
    for (int nt = 0; nt < 4; ++nt) {
      v16bf bx = load_b_frag(Ol[cur], nt * 16 + ln, hf);
      acc[0][nt] = wmma_bf16(a0, bx, acc[0][nt]);
      acc[1][nt] = wmma_bf16(a1, bx, acc[1][nt]);
    }

    stage_fence_async();
    __syncthreads();
  }

#pragma unroll
  for (int mi = 0; mi < 2; ++mi) {
#pragma unroll
    for (int r = 0; r < 8; ++r) {
      int co = (wave + mi * 8) * 16 + r + 8 * hf;
      float bv = Wbias[co];
#pragma unroll
      for (int nt = 0; nt < 4; ++nt) {
        size_t idx = ((size_t)b * 256 + co) * 1024 + hw0 + nt * 16 + ln;
        y[idx] = acc[mi][nt][r] + bv + x[idx];
      }
    }
  }
}

// ---------------------------------------------------------------------------
extern "C" void kernel_launch(void* const* d_in, const int* in_sizes, int n_in,
                              void* d_out, int out_size, void* d_ws, size_t ws_size,
                              hipStream_t stream) {
  (void)in_sizes; (void)n_in; (void)out_size; (void)ws_size;

  const float* x  = (const float*)d_in[0];
  const float* tw = (const float*)d_in[1];
  const float* tb = (const float*)d_in[2];
  const float* pw = (const float*)d_in[3];
  const float* pb = (const float*)d_in[4];
  const float* gw = (const float*)d_in[5];
  const float* gb = (const float*)d_in[6];
  const float* Ww = (const float*)d_in[7];
  const float* Wb = (const float*)d_in[8];

  bf16_t* xsw = (bf16_t*)d_ws;                       // 8*256*1024
  bf16_t* twb = xsw + (size_t)8 * 256 * 1024;        // 128*256
  bf16_t* pwb = twb + 128 * 256;
  bf16_t* gwb = pwb + 128 * 256;
  bf16_t* Wwb = gwb + 128 * 256;                     // 256*128
  bf16_t* Tm  = Wwb + 256 * 128;                     // 8192*128
  bf16_t* Psw = Tm  + (size_t)8192 * 128;
  bf16_t* Gsw = Psw + (size_t)8192 * 128;
  bf16_t* Osw = Gsw + (size_t)8192 * 128;

  cvt_x_swz_kernel<<<8192, 256, 0, stream>>>(x, xsw);
  cvt_bf16_kernel<<<128, 256, 0, stream>>>(tw, twb, 128 * 256);
  cvt_bf16_kernel<<<128, 256, 0, stream>>>(pw, pwb, 128 * 256);
  cvt_bf16_kernel<<<128, 256, 0, stream>>>(gw, gwb, 128 * 256);
  cvt_bf16_kernel<<<128, 256, 0, stream>>>(Ww, Wwb, 256 * 128);

  qkv_kernel<<<128, 256, 0, stream>>>(xsw, twb, pwb, gwb, tb, pb, gb, Tm, Psw, Gsw);
  flash_kernel<<<64, 256, 0, stream>>>(Tm, Psw, Gsw, Osw);
  outconv_kernel<<<128, 256, 0, stream>>>(Osw, Wwb, Wb, x, (float*)d_out);
}
